// QuantizedMultiHeadAttention_51127290691845
// MI455X (gfx1250) — 16-bit and 8-bit matrix operand order repaired and checked by reading, compiled, not run on hardware
//
#include <hip/hip_runtime.h>
#include <hip/hip_bf16.h>


typedef int      v8i  __attribute__((ext_vector_type(8)));
typedef float    v8f  __attribute__((ext_vector_type(8)));
typedef _Float16 v16h __attribute__((ext_vector_type(16)));

union FragH { v16h v; unsigned u[8]; };
union FragI { v8i  v; unsigned u[8]; };

__device__ __forceinline__ int kb8A(int i, int half) {
  return ((i >> 2) << 5) + (((i >> 1) & 1) << 4) + ((i & 1) << 2) + (half << 3);
}
__device__ __forceinline__ int kb8B(int i, int grp) {
  return ((i >> 2) << 5) + (grp << 4) + ((i & 3) << 2);
}
__device__ __forceinline__ int kb16A(int i, int half) {
  return (half << 4) + (i << 1);
}

__global__ __launch_bounds__(256) void quant_rows_kernel(
    const float* __restrict__ X, unsigned char* __restrict__ Q,
    int* __restrict__ rowsum, const float* __restrict__ sPtr,
    const float* __restrict__ zpPtr, int perRow, int K) {
  const int row = blockIdx.x;
  const float s  = perRow ? sPtr[row]  : sPtr[0];
  const float zp = perRow ? zpPtr[row] : zpPtr[0];
  const float inv = 1.0f / s;
  int sum = 0;
  for (int k = threadIdx.x; k < K; k += blockDim.x) {
    float q = X[(size_t)row * K + k] * inv + zp;
    q = fminf(fmaxf(q, 0.0f), 255.0f);
    int qi = (int)rintf(q);
    Q[(size_t)row * K + k] = (unsigned char)qi;
    sum += qi;
  }
  __shared__ int red[256];
  red[threadIdx.x] = sum;
  __syncthreads();
  for (int off = 128; off > 0; off >>= 1) {
    if (threadIdx.x < off) red[threadIdx.x] += red[threadIdx.x + off];
    __syncthreads();
  }
  if (threadIdx.x == 0) rowsum[row] = red[0];
}

__global__ __launch_bounds__(32) void gemm_i8_kernel(
    const unsigned char* __restrict__ A,
    const unsigned char* __restrict__ Bw,
    const int* __restrict__ rsA, const int* __restrict__ rsB,
    const float* __restrict__ sA, const float* __restrict__ zpA,
    const float* __restrict__ sW, const float* __restrict__ zpW,
    const float* __restrict__ bias, float* __restrict__ out,
    _Float16* __restrict__ qh, _Float16* __restrict__ kh,
    _Float16* __restrict__ vT, int pack, int M, int N, int K) {
  const int lane = threadIdx.x;
  const int lm   = lane & 15;
  const int half = lane >> 4;
  const int m0 = blockIdx.y * 32;
  const int n0 = blockIdx.x * 64;

  v8i acc[2][4];
#pragma unroll
  for (int mt = 0; mt < 2; ++mt)
#pragma unroll
    for (int nt = 0; nt < 4; ++nt) acc[mt][nt] = (v8i)0;

  for (int k = 0; k < K; k += 64) {
    FragI a[2];
#pragma unroll
    for (int mt = 0; mt < 2; ++mt) {
      const unsigned char* arow = A + (size_t)(m0 + mt * 16 + lm) * K + k;
#pragma unroll
      for (int i = 0; i < 8; ++i)
        a[mt].u[i] = *(const unsigned*)(arow + kb8A(i, half));
    }
#pragma unroll
    for (int nt = 0; nt < 4; ++nt) {
      FragI b;
      const unsigned char* brow = Bw + (size_t)(n0 + nt * 16 + lm) * K + k;
#pragma unroll
      for (int i = 0; i < 8; ++i)
        b.u[i] = *(const unsigned*)(brow + kb8A(i, half));
      acc[0][nt] = __builtin_amdgcn_wmma_i32_16x16x64_iu8(
          false, a[0].v, false, b.v, acc[0][nt], false, false);
      acc[1][nt] = __builtin_amdgcn_wmma_i32_16x16x64_iu8(
          false, a[1].v, false, b.v, acc[1][nt], false, false);
    }
  }

  const float sa = sA[0], za = zpA[0];
#pragma unroll
  for (int nt = 0; nt < 4; ++nt) {
    const int col = n0 + nt * 16 + lm;
    const float zw = zpW[col], sw = sW[col];
    const float rb = (float)rsB[col];
    const float bi = bias[col];
    const float cK = (float)K * za * zw;
#pragma unroll
    for (int mt = 0; mt < 2; ++mt) {
#pragma unroll
      for (int r = 0; r < 8; ++r) {
        const int row = m0 + mt * 16 + r + half * 8;
        float v = (float)acc[mt][nt][r] - zw * (float)rsA[row] - za * rb + cK;
        const float val = sa * sw * v + bi;
        if (!pack) {
          out[(size_t)row * N + col] = val;
        } else if (col < 1024) {
          qh[(size_t)row * 1024 + col] = (_Float16)(val * 0.125f);
        } else if (col < 2048) {
          kh[(size_t)row * 1024 + (col - 1024)] = (_Float16)val;
        } else {
          const int e = col - 2048;
          const int hh = e >> 6, d = e & 63;
          const int bb = row >> 10, s = row & 1023;
          vT[(size_t)(((bb << 4) + hh) * 64 + d) * 1024 + s] = (_Float16)val;
        }
      }
    }
  }
}

__global__ __launch_bounds__(32) void attn_kernel(
    const _Float16* __restrict__ qh,
    const _Float16* __restrict__ kh,
    const _Float16* __restrict__ vT,
    float* __restrict__ attn_out,
    const float* __restrict__ sAttnP, const float* __restrict__ zpAttnP) {
  const int lane = threadIdx.x;
  const int lm   = lane & 15;
  const int half = lane >> 4;
  const int bh = blockIdx.x;
  const int b = bh >> 4, h = bh & 15;
  const int q0 = blockIdx.y * 16;
  const float sAttn = sAttnP[0], zpAttn = zpAttnP[0], invS = 1.0f / sAttnP[0];

  __shared__ _Float16 sP[16][32];

  FragH aq[2];
  {
    const _Float16* qrow = qh + (size_t)(b * 1024 + q0 + lm) * 1024 + h * 64;
#pragma unroll
    for (int c = 0; c < 2; ++c)
#pragma unroll
      for (int i = 0; i < 8; ++i)
        aq[c].u[i] = *(const unsigned*)&qrow[c * 32 + kb16A(i, half)];
  }

  float mrun[8], lrun[8];
  v8f o[4];
#pragma unroll
  for (int r = 0; r < 8; ++r) { mrun[r] = -1e30f; lrun[r] = 0.0f; }
#pragma unroll
  for (int t = 0; t < 4; ++t) o[t] = (v8f)0.0f;

  for (int t0 = 0; t0 < 1024; t0 += 32) {
    v8f sc[2];
#pragma unroll
    for (int sub = 0; sub < 2; ++sub) {
      const _Float16* krow = kh +
          (size_t)(b * 1024 + t0 + sub * 16 + lm) * 1024 + h * 64 + half * 16;
      FragH bk0, bk1;
#pragma unroll
      for (int i = 0; i < 8; ++i) {
        bk0.u[i] = *(const unsigned*)&krow[2 * i];
        bk1.u[i] = *(const unsigned*)&krow[32 + 2 * i];
      }
      v8f z = (v8f)0.0f;
      z = __builtin_amdgcn_wmma_f32_16x16x32_f16(false, aq[0].v, false, bk0.v,
                                                 (short)0, z, false, false);
      z = __builtin_amdgcn_wmma_f32_16x16x32_f16(false, aq[1].v, false, bk1.v,
                                                 (short)0, z, false, false);
#pragma unroll
      for (int r = 0; r < 8; ++r) {
        float q = fminf(fmaxf(z[r] * invS + zpAttn, 0.0f), 255.0f);
        z[r] = (rintf(q) - zpAttn) * sAttn;
      }
      sc[sub] = z;
    }
    float p0[8], p1[8];
#pragma unroll
    for (int r = 0; r < 8; ++r) {
      float rm = fmaxf(sc[0][r], sc[1][r]);
#pragma unroll
      for (int msk = 8; msk >= 1; msk >>= 1)
        rm = fmaxf(rm, __shfl_xor(rm, msk, 32));
      const float newm = fmaxf(mrun[r], rm);
      const float alpha = __expf(mrun[r] - newm);
      p0[r] = __expf(sc[0][r] - newm);
      p1[r] = __expf(sc[1][r] - newm);
      float rs = p0[r] + p1[r];
#pragma unroll
      for (int msk = 8; msk >= 1; msk >>= 1)
        rs += __shfl_xor(rs, msk, 32);
      lrun[r] = lrun[r] * alpha + rs;
      mrun[r] = newm;
#pragma unroll
      for (int t = 0; t < 4; ++t) o[t][r] *= alpha;
    }
#pragma unroll
    for (int r = 0; r < 8; ++r) {
      sP[r + half * 8][lm]      = (_Float16)p0[r];
      sP[r + half * 8][16 + lm] = (_Float16)p1[r];
    }
    __syncthreads();
    FragH pf;
#pragma unroll
    for (int i = 0; i < 8; ++i)
      pf.u[i] = *(const unsigned*)&sP[lm][kb16A(i, half)];
    __syncthreads();
#pragma unroll
    for (int dc = 0; dc < 4; ++dc) {
      const _Float16* vrow = vT +
          (size_t)(bh * 64 + dc * 16 + lm) * 1024 + t0 + half * 16;
      FragH vf;
#pragma unroll
      for (int i = 0; i < 8; ++i)
        vf.u[i] = *(const unsigned*)&vrow[2 * i];
      o[dc] = __builtin_amdgcn_wmma_f32_16x16x32_f16(false, pf.v, false, vf.v,
                                                     (short)0, o[dc], false, false);
    }
  }
#pragma unroll
  for (int dc = 0; dc < 4; ++dc)
#pragma unroll
    for (int r = 0; r < 8; ++r) {
      const int tok = b * 1024 + q0 + r + half * 8;
      attn_out[(size_t)tok * 1024 + h * 64 + dc * 16 + lm] =
          o[dc][r] / lrun[r];
    }
}

extern "C" void kernel_launch(void* const* d_in, const int* in_sizes, int n_in,
                              void* d_out, int out_size, void* d_ws,
                              size_t ws_size, hipStream_t stream) {
  (void)in_sizes; (void)n_in; (void)out_size; (void)ws_size;
  const float* x        = (const float*)d_in[0];
  const float* W_qkv    = (const float*)d_in[1];
  const float* b_qkv    = (const float*)d_in[2];
  const float* W_proj   = (const float*)d_in[3];
  const float* b_proj   = (const float*)d_in[4];
  const float* s_a_qkv  = (const float*)d_in[5];
  const float* zp_a_qkv = (const float*)d_in[6];
  const float* s_w_qkv  = (const float*)d_in[7];
  const float* zp_w_qkv = (const float*)d_in[8];
  const float* s_attn   = (const float*)d_in[9];
  const float* zp_attn  = (const float*)d_in[10];
  const float* s_a_prj  = (const float*)d_in[11];
  const float* zp_a_prj = (const float*)d_in[12];
  const float* s_w_prj  = (const float*)d_in[13];
  const float* zp_w_prj = (const float*)d_in[14];
  float* out = (float*)d_out;

  char* ws = (char*)d_ws;
  unsigned char* qx1 = (unsigned char*)(ws);
  unsigned char* qw1 = (unsigned char*)(ws + (4u  << 20));
  int*   rsx1        = (int*)          (ws + (7u  << 20));
  int*   rsw1        = (int*)          (ws + (7u  << 20) + (64u << 10));
  _Float16* qh       = (_Float16*)     (ws + (8u  << 20));
  _Float16* kh       = (_Float16*)     (ws + (16u << 20));
  _Float16* vT       = (_Float16*)     (ws + (24u << 20));
  float* attno       = (float*)        (ws + (32u << 20));
  unsigned char* qx2 = (unsigned char*)(ws + (48u << 20));
  unsigned char* qw2 = (unsigned char*)(ws + (52u << 20));
  int*   rsx2        = (int*)          (ws + (53u << 20));
  int*   rsw2        = (int*)          (ws + (53u << 20) + (64u << 10));

  quant_rows_kernel<<<4096, 256, 0, stream>>>(x, qx1, rsx1, s_a_qkv, zp_a_qkv, 0, 1024);
  quant_rows_kernel<<<3072, 256, 0, stream>>>(W_qkv, qw1, rsw1, s_w_qkv, zp_w_qkv, 1, 1024);
  gemm_i8_kernel<<<dim3(3072 / 64, 4096 / 32), 32, 0, stream>>>(
      qx1, qw1, rsx1, rsw1, s_a_qkv, zp_a_qkv, s_w_qkv, zp_w_qkv, b_qkv,
      nullptr, qh, kh, vT, 1, 4096, 3072, 1024);
  attn_kernel<<<dim3(64, 64), 32, 0, stream>>>(qh, kh, vT, attno, s_attn, zp_attn);
  quant_rows_kernel<<<4096, 256, 0, stream>>>(attno, qx2, rsx2, s_a_prj, zp_a_prj, 0, 1024);
  quant_rows_kernel<<<1024, 256, 0, stream>>>(W_proj, qw2, rsw2, s_w_prj, zp_w_prj, 1, 1024);
  gemm_i8_kernel<<<dim3(1024 / 64, 4096 / 32), 32, 0, stream>>>(
      qx2, qw2, rsx2, rsw2, s_a_prj, zp_a_prj, s_w_prj, zp_w_prj, b_proj,
      out, nullptr, nullptr, nullptr, 0, 4096, 1024, 1024);
}
